// MultiHeadCrossAttention_23141283791475
// MI455X (gfx1250) — compile-verified, matrix operand order checked by reading, not run on hardware
//
#include <hip/hip_runtime.h>
#include <hip/hip_bf16.h>

typedef __attribute__((ext_vector_type(16))) __bf16 v16bf;
typedef __attribute__((ext_vector_type(8)))  __bf16 v8bf;
typedef __attribute__((ext_vector_type(8)))  float  v8f;

#define B_  2
#define H_  16
#define N_  2048
#define D_  1024
#define DH_ 64

static __device__ __forceinline__ v16bf cat16(v8bf lo, v8bf hi) {
  v16bf r;
#pragma unroll
  for (int i = 0; i < 8; ++i) { r[i] = lo[i]; r[i + 8] = hi[i]; }
  return r;
}

static __device__ __forceinline__ v8bf ld8(const __bf16* p) {
  return *(const v8bf*)p;
}

static __device__ __forceinline__ v8f wmma_bf16(v16bf a, v16bf b, v8f c) {
  return __builtin_amdgcn_wmma_f32_16x16x32_bf16(false, a, false, b, (short)0, c,
                                                 false, false);
}

// ---------------------------------------------------------------------------
// K0: transpose 1024x1024 fp32 weight -> bf16 [N,K] (row over contraction dim)
// ---------------------------------------------------------------------------
__global__ void wt_transpose(const float* __restrict__ W, __bf16* __restrict__ Wt) {
  __shared__ float t[32][33];
  const int tx = threadIdx.x, ty = threadIdx.y;
  const int x0 = blockIdx.x * 32, y0 = blockIdx.y * 32;
#pragma unroll
  for (int j = 0; j < 4; ++j)
    t[ty + j * 8][tx] = W[(size_t)(y0 + ty + j * 8) * D_ + x0 + tx];
  __syncthreads();
#pragma unroll
  for (int j = 0; j < 4; ++j)
    Wt[(size_t)(x0 + ty + j * 8) * D_ + y0 + tx] = (__bf16)t[tx][ty + j * 8];
}

// ---------------------------------------------------------------------------
// K1: projection GEMM  out = X @ W + bias   (X fp32 [4096,1024], Wt bf16 [N,K])
// One wave per 16(M) x 64(N) tile; K=1024 in steps of 32 (bf16 WMMA).
// transposed==0: out bf16 [B,H,N,64];  transposed==1: out bf16 [B,H,64,N] (V^T)
// ---------------------------------------------------------------------------
__global__ void proj_gemm(const float* __restrict__ X, const __bf16* __restrict__ Wt,
                          const float* __restrict__ bias, __bf16* __restrict__ out,
                          int transposed) {
  const int lane = threadIdx.x;
  const int m = lane & 15, hf = lane >> 4;
  const int m0 = blockIdx.x * 16;     // global row tile (B*N rows)
  const int n0 = blockIdx.y * 64;     // output column tile
  const int b  = m0 >> 11;            // batch
  const int rl = m0 & (N_ - 1);       // row within batch

  const float* Xrow = X + (size_t)(m0 + m) * D_;

  v8f acc[4] = {{}, {}, {}, {}};
  for (int k0 = 0; k0 < D_; k0 += 32) {
    const float* ap = Xrow + k0 + hf * 8;
    float4 l0 = *(const float4*)(ap);
    float4 l1 = *(const float4*)(ap + 4);
    float4 h0 = *(const float4*)(ap + 16);
    float4 h1 = *(const float4*)(ap + 20);
    v16bf a;
    a[0] = (__bf16)l0.x; a[1] = (__bf16)l0.y; a[2] = (__bf16)l0.z; a[3] = (__bf16)l0.w;
    a[4] = (__bf16)l1.x; a[5] = (__bf16)l1.y; a[6] = (__bf16)l1.z; a[7] = (__bf16)l1.w;
    a[8]  = (__bf16)h0.x; a[9]  = (__bf16)h0.y; a[10] = (__bf16)h0.z; a[11] = (__bf16)h0.w;
    a[12] = (__bf16)h1.x; a[13] = (__bf16)h1.y; a[14] = (__bf16)h1.z; a[15] = (__bf16)h1.w;
#pragma unroll
    for (int g = 0; g < 4; ++g) {
      const __bf16* wp = Wt + (size_t)(n0 + g * 16 + m) * D_ + k0 + hf * 8;
      v16bf bf = cat16(ld8(wp), ld8(wp + 16));
      acc[g] = wmma_bf16(a, bf, acc[g]);
    }
  }

#pragma unroll
  for (int g = 0; g < 4; ++g) {
    const int col = n0 + g * 16 + m;
    const float bv = bias[col];
    const int h = col >> 6, d = col & 63;
    if (!transposed) {
#pragma unroll
      for (int i = 0; i < 8; ++i)
        out[((size_t)(b * H_ + h) * N_ + rl + i + hf * 8) * DH_ + d] =
            (__bf16)(acc[g][i] + bv);
    } else {
      v8bf pk;
#pragma unroll
      for (int i = 0; i < 8; ++i) pk[i] = (__bf16)(acc[g][i] + bv);
      *(v8bf*)(out + ((size_t)(b * H_ + h) * DH_ + d) * N_ + rl + hf * 8) = pk;
    }
  }
}

// ---------------------------------------------------------------------------
// K2: attention. One wave per (b,h,16-row q tile).
// Pass 1: streaming QK^T with online row max/sum (shfl_xor over 16-lane halves
// matching the WMMA C layout). Pass 2: recompute scores (K panel hot in L2),
// write normalized attn fp32, transpose 16x32 prob tile via LDS into the WMMA
// A layout, accumulate context = P @ V with V^T-resident B fragments.
// ---------------------------------------------------------------------------
__global__ void attention(const __bf16* __restrict__ Qb, const __bf16* __restrict__ Kb,
                          const __bf16* __restrict__ Vt, float* __restrict__ ctx,
                          float* __restrict__ attn) {
  __shared__ __bf16 Pld[32][17];
  const int lane = threadIdx.x;
  const int m = lane & 15, hf = lane >> 4;
  const int q0 = blockIdx.x * 16;
  const int h = blockIdx.y, b = blockIdx.z;
  const size_t bh = (size_t)(b * H_ + h);

  const __bf16* qp = Qb + (bh * N_ + q0 + m) * DH_ + hf * 8;
  const v16bf aQ0 = cat16(ld8(qp), ld8(qp + 16));
  const v16bf aQ1 = cat16(ld8(qp + 32), ld8(qp + 48));

  const __bf16* kbase = Kb + bh * N_ * DH_;
  const __bf16* vbase = Vt + bh * DH_ * N_;

  float mrun[8], srun[8];
#pragma unroll
  for (int i = 0; i < 8; ++i) { mrun[i] = -1e30f; srun[i] = 0.f; }

  // ---- pass 1: online softmax statistics ----
  for (int kt = 0; kt < N_ / 16; ++kt) {
    const __bf16* kp = kbase + (size_t)(kt * 16 + m) * DH_ + hf * 8;
    v16bf bK0 = cat16(ld8(kp), ld8(kp + 16));
    v16bf bK1 = cat16(ld8(kp + 32), ld8(kp + 48));
    v8f acc = {};
    acc = wmma_bf16(aQ0, bK0, acc);
    acc = wmma_bf16(aQ1, bK1, acc);
#pragma unroll
    for (int i = 0; i < 8; ++i) {
      float s = acc[i] * 0.125f;        // 1/sqrt(64)
      float tm = s;
      tm = fmaxf(tm, __shfl_xor(tm, 1, 16));
      tm = fmaxf(tm, __shfl_xor(tm, 2, 16));
      tm = fmaxf(tm, __shfl_xor(tm, 4, 16));
      tm = fmaxf(tm, __shfl_xor(tm, 8, 16));
      float mn = fmaxf(mrun[i], tm);
      float e = __expf(s - mn);
      float ts = e;
      ts += __shfl_xor(ts, 1, 16);
      ts += __shfl_xor(ts, 2, 16);
      ts += __shfl_xor(ts, 4, 16);
      ts += __shfl_xor(ts, 8, 16);
      srun[i] = srun[i] * __expf(mrun[i] - mn) + ts;
      mrun[i] = mn;
    }
  }
  float rinv[8];
#pragma unroll
  for (int i = 0; i < 8; ++i) rinv[i] = 1.f / srun[i];

  // ---- pass 2: write attn, accumulate context ----
  v8f cacc[4] = {{}, {}, {}, {}};
  for (int ktp = 0; ktp < N_ / 32; ++ktp) {
#pragma unroll
    for (int sub = 0; sub < 2; ++sub) {
      const int kt = ktp * 2 + sub;
      const __bf16* kp = kbase + (size_t)(kt * 16 + m) * DH_ + hf * 8;
      v16bf bK0 = cat16(ld8(kp), ld8(kp + 16));
      v16bf bK1 = cat16(ld8(kp + 32), ld8(kp + 48));
      v8f acc = {};
      acc = wmma_bf16(aQ0, bK0, acc);
      acc = wmma_bf16(aQ1, bK1, acc);
#pragma unroll
      for (int i = 0; i < 8; ++i) {
        float p = __expf(acc[i] * 0.125f - mrun[i]) * rinv[i];
        attn[(bh * N_ + q0 + i + hf * 8) * (size_t)N_ + kt * 16 + m] = p;
        Pld[sub * 16 + m][i + hf * 8] = (__bf16)p;
      }
    }
    __syncthreads();
    v16bf aP;
#pragma unroll
    for (int e = 0; e < 8; ++e) {
      aP[e]     = Pld[hf * 8 + e][m];
      aP[e + 8] = Pld[hf * 8 + 16 + e][m];
    }
#pragma unroll
    for (int g = 0; g < 4; ++g) {
      const __bf16* vp = vbase + (size_t)(g * 16 + m) * N_ + ktp * 32 + hf * 8;
      v16bf bV = cat16(ld8(vp), ld8(vp + 16));
      cacc[g] = wmma_bf16(aP, bV, cacc[g]);
    }
    __syncthreads();
  }

#pragma unroll
  for (int g = 0; g < 4; ++g)
#pragma unroll
    for (int i = 0; i < 8; ++i)
      ctx[(size_t)(b * N_ + q0 + i + hf * 8) * (H_ * DH_) + h * DH_ + g * 16 + m] =
          cacc[g][i];
}

// ---------------------------------------------------------------------------
extern "C" void kernel_launch(void* const* d_in, const int* in_sizes, int n_in,
                              void* d_out, int out_size, void* d_ws, size_t ws_size,
                              hipStream_t stream) {
  const float* query = (const float*)d_in[0];
  const float* key   = (const float*)d_in[1];
  const float* value = (const float*)d_in[2];
  const float* Wq    = (const float*)d_in[3];
  const float* bq    = (const float*)d_in[4];
  const float* Wk    = (const float*)d_in[5];
  const float* bk    = (const float*)d_in[6];
  const float* Wv    = (const float*)d_in[7];
  const float* bv    = (const float*)d_in[8];

  char* ws = (char*)d_ws;
  const size_t WT_BYTES = (size_t)D_ * D_ * sizeof(__bf16);            // 2 MB each
  const size_t PROJ_BYTES = (size_t)B_ * H_ * N_ * DH_ * sizeof(__bf16); // 8 MB each
  __bf16* Wt_q = (__bf16*)(ws);
  __bf16* Wt_k = (__bf16*)(ws + WT_BYTES);
  __bf16* Wt_v = (__bf16*)(ws + 2 * WT_BYTES);
  __bf16* Qb   = (__bf16*)(ws + 3 * WT_BYTES);
  __bf16* Kb   = (__bf16*)(ws + 3 * WT_BYTES + PROJ_BYTES);
  __bf16* Vt   = (__bf16*)(ws + 3 * WT_BYTES + 2 * PROJ_BYTES);

  dim3 tb(32, 8), tg(D_ / 32, D_ / 32);
  wt_transpose<<<tg, tb, 0, stream>>>(Wq, Wt_q);
  wt_transpose<<<tg, tb, 0, stream>>>(Wk, Wt_k);
  wt_transpose<<<tg, tb, 0, stream>>>(Wv, Wt_v);

  dim3 pg(B_ * N_ / 16, D_ / 64);
  proj_gemm<<<pg, 32, 0, stream>>>(query, Wt_q, bq, Qb, 0);
  proj_gemm<<<pg, 32, 0, stream>>>(key,   Wt_k, bk, Kb, 0);
  proj_gemm<<<pg, 32, 0, stream>>>(value, Wt_v, bv, Vt, 1);

  float* ctx  = (float*)d_out;
  float* attn = (float*)d_out + (size_t)B_ * N_ * H_ * DH_;
  dim3 ag(N_ / 16, H_, B_);
  attention<<<ag, 32, 0, stream>>>(Qb, Kb, Vt, ctx, attn);
}
